// DFDGCN_14259291422898
// MI455X (gfx1250) — matrix operand order repaired and checked by reading, not compiled, not run on hardware
//
#include <hip/hip_runtime.h>
#include <hip/hip_bf16.h>
#include <cstddef>
#include <cstdint>

// ---------------------------------------------------------------------------
// Model dims
// ---------------------------------------------------------------------------
#define NB   16      // batch
#define NL   12      // seq len
#define NN   512     // nodes
#define NRC  32      // residual channels
#define NDC  32      // dilation channels
#define NSC  256     // skip channels
#define NEC  512     // end channels
#define NOUT 12
#define NEMB 152     // 64 + 64 + 12 + 12
#define NEMBP 160    // padded to 5*32
#define NHID 128
#define NRECEPT 13

typedef __attribute__((ext_vector_type(16))) __bf16 bf16x16;
typedef __attribute__((ext_vector_type(8)))  float  f32x8;

// ---------------------------------------------------------------------------
// WMMA fragment helpers (wave32, gfx1250 layouts, CDNA5 ISA 7.12.2)
// A: 16x32 bf16 (MxK).  lanes 0-15: M=lane; V0..3 hold K=0..7, V4..7 K=16..23
//                        lanes16-31: same M, K offset +8 (K=8..15 / 24..31)
// B: 32x16 bf16 (KxN).  n=l&15; K layout same as A (K=0..7/16..23, +8 for lanes16-31)
// C/D: f32 16x16.       reg r, lane l: M = r + 8*(l>>4), N = l&15
// ---------------------------------------------------------------------------
template <typename F>
static __device__ inline bf16x16 load_a_frag(F getA) {  // getA -> float
  int lane = threadIdx.x & 31;
  int m = lane & 15, half = lane >> 4;
  bf16x16 a;
#pragma unroll
  for (int j = 0; j < 8; ++j) {
    int k0 = (j < 4) ? (2 * j + 8 * half) : (16 + 2 * (j - 4) + 8 * half);
    a[2 * j]     = (__bf16)getA(m, k0);
    a[2 * j + 1] = (__bf16)getA(m, k0 + 1);
  }
  return a;
}

template <typename F>
static __device__ inline bf16x16 load_a_frag_bf(F getA) {  // getA -> __bf16
  int lane = threadIdx.x & 31;
  int m = lane & 15, half = lane >> 4;
  bf16x16 a;
#pragma unroll
  for (int j = 0; j < 8; ++j) {
    int k0 = (j < 4) ? (2 * j + 8 * half) : (16 + 2 * (j - 4) + 8 * half);
    a[2 * j]     = getA(m, k0);
    a[2 * j + 1] = getA(m, k0 + 1);
  }
  return a;
}

template <typename F>
static __device__ inline bf16x16 load_b_frag(F getB) {  // getB -> float
  int lane = threadIdx.x & 31;
  int n = lane & 15, half = lane >> 4;
  bf16x16 b;
#pragma unroll
  for (int j = 0; j < 8; ++j) {
    int k0 = (j < 4) ? (2 * j + 8 * half) : (16 + 2 * (j - 4) + 8 * half);
    b[2 * j]     = (__bf16)getB(k0, n);
    b[2 * j + 1] = (__bf16)getB(k0 + 1, n);
  }
  return b;
}

template <typename F>
static __device__ inline bf16x16 load_b_frag_bf(F getB) {  // getB -> __bf16
  int lane = threadIdx.x & 31;
  int n = lane & 15, half = lane >> 4;
  bf16x16 b;
#pragma unroll
  for (int j = 0; j < 8; ++j) {
    int k0 = (j < 4) ? (2 * j + 8 * half) : (16 + 2 * (j - 4) + 8 * half);
    b[2 * j]     = getB(k0, n);
    b[2 * j + 1] = getB(k0 + 1, n);
  }
  return b;
}

#define WMMA_BF16(acc, a, b)                                                   \
  acc = __builtin_amdgcn_wmma_f32_16x16x32_bf16(false, a, false, b, (short)0,  \
                                                acc, false, false)

// ---------------------------------------------------------------------------
// Utility kernels
// ---------------------------------------------------------------------------
__global__ void k_zero(float* p, int n) {
  int i = blockIdx.x * blockDim.x + threadIdx.x;
  if (i < n) p[i] = 0.f;
}

// pack tconv weights -> bf16 [layer][tap][o][k] (k contiguous per out chan)
__global__ void k_prep_tw(const float* __restrict__ fw,
                          const float* __restrict__ gw,
                          __bf16* __restrict__ fwt, __bf16* __restrict__ gwt) {
  int idx = blockIdx.x * blockDim.x + threadIdx.x;  // 8*2*32*32
  if (idx >= 8 * 2 * NDC * NRC) return;
  int k = idx & 31;
  int o = (idx >> 5) & 31;
  int tap = (idx >> 10) & 1;
  int i = idx >> 11;
  fwt[idx] = (__bf16)fw[(((size_t)i * NDC + o) * NRC + k) * 2 + tap];
  gwt[idx] = (__bf16)gw[(((size_t)i * NDC + o) * NRC + k) * 2 + tap];
}

// Wxabs transposed -> bf16 [j][k]
__global__ void k_prep_wx(const float* __restrict__ Wx,
                          __bf16* __restrict__ Wxt) {
  int idx = blockIdx.x * blockDim.x + threadIdx.x;  // 128*128
  if (idx >= NHID * NHID) return;
  int k = idx & (NHID - 1), j = idx >> 7;
  Wxt[(size_t)j * NHID + k] = (__bf16)Wx[(size_t)k * NHID + j];
}

// Wd transposed+padded -> bf16 [n][j][k160]  (k contiguous, zero padded)
__global__ void k_prep_wd(const float* __restrict__ Wd,
                          __bf16* __restrict__ Wdt) {
  int idx = blockIdx.x * blockDim.x + threadIdx.x;  // 512*128*160
  if (idx >= NN * NHID * NEMBP) return;
  int k = idx % NEMBP;
  int j = (idx / NEMBP) & (NHID - 1);
  int n = idx / (NEMBP * NHID);
  Wdt[idx] =
      (k < NEMB) ? (__bf16)Wd[((size_t)n * NEMB + k) * NHID + j] : (__bf16)0.f;
}

// start conv: x0[b][l][c][n], l in 0..12 (left-pad 1), 2->32 channels
__global__ void k_start(const float* __restrict__ hd,
                        const float* __restrict__ sw,
                        const float* __restrict__ sb,
                        float* __restrict__ x, __bf16* __restrict__ xbf) {
  int idx = blockIdx.x * blockDim.x + threadIdx.x;  // over b*13*512
  if (idx >= NB * NRECEPT * NN) return;
  int n = idx & (NN - 1);
  int l = (idx >> 9) % NRECEPT;
  int b = idx / (NRECEPT * NN);
  float in0 = 0.f, in1 = 0.f;
  if (l >= 1) {
    const float* p = hd + (((size_t)b * NL + (l - 1)) * NN + n) * 3;
    in0 = p[0];
    in1 = p[1];
  }
#pragma unroll 4
  for (int c = 0; c < NRC; ++c) {
    float v = sw[c * 2 + 0] * in0 + sw[c * 2 + 1] * in1 + sb[c];
    size_t o = (((size_t)b * NRECEPT + l) * NRC + c) * NN + n;
    x[o] = v;
    xbf[o] = (__bf16)v;
  }
}

// gwadp = row-softmax(relu(nv1 @ nv2)); stored TRANSPOSED bf16:
// outT[w][v] = softmax_row_v(col w).  1 block / row v.
__global__ void k_gwadp(const float* __restrict__ nv1,
                        const float* __restrict__ nv2,
                        __bf16* __restrict__ outT) {
  __shared__ float rowv[NN];
  __shared__ float sred[256];
  int i = blockIdx.x, tid = threadIdx.x;
#pragma unroll
  for (int jj = 0; jj < 2; ++jj) {
    int j = tid + jj * 256;
    float acc = 0.f;
    for (int k = 0; k < 64; ++k) acc += nv1[i * 64 + k] * nv2[k * NN + j];
    rowv[j] = fmaxf(acc, 0.f);
  }
  __syncthreads();
  float m = fmaxf(rowv[tid], rowv[tid + 256]);
  sred[tid] = m;
  __syncthreads();
  for (int s = 128; s > 0; s >>= 1) {
    if (tid < s) sred[tid] = fmaxf(sred[tid], sred[tid + s]);
    __syncthreads();
  }
  float rmax = sred[0];
  __syncthreads();
  float e0 = __expf(rowv[tid] - rmax), e1 = __expf(rowv[tid + 256] - rmax);
  sred[tid] = e0 + e1;
  __syncthreads();
  for (int s = 128; s > 0; s >>= 1) {
    if (tid < s) sred[tid] += sred[tid + s];
    __syncthreads();
  }
  float inv = 1.f / sred[0];
  outT[(size_t)tid * NN + i] = (__bf16)(e0 * inv);
  outT[(size_t)(tid + 256) * NN + i] = (__bf16)(e1 * inv);
}

// e_bf[b][n][160] (bf16, zero padded): |rfft12|@Ex1 |node1 |TiD |DiW
__global__ void k_embed(const float* __restrict__ hd,
                        const float* __restrict__ Ex1,
                        const float* __restrict__ node1,
                        const float* __restrict__ tid_emb,
                        const float* __restrict__ diw_emb,
                        __bf16* __restrict__ e_bf) {
  int idx = blockIdx.x * blockDim.x + threadIdx.x;  // b*n
  if (idx >= NB * NN) return;
  int n = idx & (NN - 1), b = idx >> 9;
  float xt[NL];
  for (int t = 0; t < NL; ++t)
    xt[t] = hd[(((size_t)b * NL + t) * NN + n) * 3 + 0];
  float freq[7];
#pragma unroll
  for (int k = 0; k < 7; ++k) {
    float re = 0.f, im = 0.f;
    for (int t = 0; t < NL; ++t) {
      float ang = -2.f * 3.14159265358979f * (float)(k * t) / 12.f;
      re += xt[t] * __cosf(ang);
      im += xt[t] * __sinf(ang);
    }
    freq[k] = sqrtf(re * re + im * im);
  }
  __bf16* ep = e_bf + ((size_t)b * NN + n) * NEMBP;
  for (int k = 0; k < 64; ++k) {
    float a = 0.f;
    for (int f = 0; f < 7; ++f) a += freq[f] * Ex1[f * 64 + k];
    ep[k] = (__bf16)a;
  }
  for (int k = 0; k < 64; ++k) ep[64 + k] = (__bf16)node1[n * 64 + k];
  float t1 = hd[(((size_t)b * NL + 11) * NN + n) * 3 + 1];
  int it = (int)(t1 * 288.f);
  it = it < 0 ? 0 : (it > 287 ? 287 : it);
  float t2 = hd[(((size_t)b * NL + 11) * NN + n) * 3 + 2];
  int id = (int)(t2 * 7.f);
  id = id < 0 ? 0 : (id > 6 ? 6 : id);
  for (int k = 0; k < NL; ++k) {
    ep[128 + k] = (__bf16)tid_emb[it * NL + k];
    ep[140 + k] = (__bf16)diw_emb[id * NL + k];
  }
  for (int k = NEMB; k < NEMBP; ++k) ep[k] = (__bf16)0.f;
}

// ---------------------------------------------------------------------------
// WMMA GEMM kernels (one wave = one 16x16 f32 tile)
// ---------------------------------------------------------------------------
// adp[b][n][k] = e[b][n][:160] @ Wdt[n] — all contiguous bf16
__global__ __launch_bounds__(32) void k_adp(const __bf16* __restrict__ e_bf,
                                            const __bf16* __restrict__ Wdt,
                                            float* __restrict__ adp) {
  int n = blockIdx.x, jt = blockIdx.y;
  f32x8 acc = {};
#pragma unroll
  for (int ks = 0; ks < 5; ++ks) {
    int kb = ks * 32;
    bf16x16 a = load_a_frag_bf([&](int m, int k) {
      return e_bf[((size_t)m * NN + n) * NEMBP + kb + k];
    });
    bf16x16 bm = load_b_frag_bf([&](int k, int j) {
      return Wdt[((size_t)n * NHID + jt * 16 + j) * NEMBP + kb + k];
    });
    WMMA_BF16(acc, a, bm);
  }
  int lane = threadIdx.x, half = lane >> 4, col = lane & 15;
#pragma unroll
  for (int r = 0; r < 8; ++r) {
    int m = r + 8 * half;  // batch
    adp[((size_t)m * NN + n) * NHID + jt * 16 + col] = acc[r];
  }
}

// per-batch layernorm of adp over (n,k); writes f32 in place + bf16 copy
__global__ void k_lnorm(float* __restrict__ adp, __bf16* __restrict__ adp_bf) {
  __shared__ float s1[256], s2[256];
  int b = blockIdx.x, tid = threadIdx.x;
  const int TOT = NN * NHID;
  float* p = adp + (size_t)b * TOT;
  float sum = 0.f, sq = 0.f;
  for (int i = tid; i < TOT; i += 256) {
    float v = p[i];
    sum += v;
    sq += v * v;
  }
  s1[tid] = sum;
  s2[tid] = sq;
  __syncthreads();
  for (int s = 128; s > 0; s >>= 1) {
    if (tid < s) {
      s1[tid] += s1[tid + s];
      s2[tid] += s2[tid + s];
    }
    __syncthreads();
  }
  float mu = s1[0] / TOT;
  float var = s2[0] / TOT - mu * mu;
  float inv = rsqrtf(var + 1e-8f);
  for (int i = tid; i < TOT; i += 256) {
    float v = (p[i] - mu) * inv;
    p[i] = v;
    adp_bf[(size_t)b * TOT + i] = (__bf16)v;
  }
}

// t[b][n][j] = adp[b][n][:] @ Wxabs   (B from transposed bf16 Wxt[j][k])
__global__ __launch_bounds__(32) void k_txw(const __bf16* __restrict__ adp_bf,
                                            const __bf16* __restrict__ Wxt,
                                            __bf16* __restrict__ t_bf) {
  int b = blockIdx.x, nt = blockIdx.y, jt = blockIdx.z;
  f32x8 acc = {};
#pragma unroll
  for (int ks = 0; ks < 4; ++ks) {
    int kb = ks * 32;
    bf16x16 a = load_a_frag_bf([&](int m, int k) {
      return adp_bf[((size_t)b * NN + nt * 16 + m) * NHID + kb + k];
    });
    bf16x16 bm = load_b_frag_bf([&](int k, int j) {
      return Wxt[(size_t)(jt * 16 + j) * NHID + kb + k];
    });
    WMMA_BF16(acc, a, bm);
  }
  int lane = threadIdx.x, half = lane >> 4, col = lane & 15;
#pragma unroll
  for (int r = 0; r < 8; ++r) {
    int m = r + 8 * half;
    t_bf[((size_t)b * NN + nt * 16 + m) * NHID + jt * 16 + col] = (__bf16)acc[r];
  }
}

// adj[b][n][m] = relu( t[b][n][:] . adp[b][m][:] )  (B rows contiguous)
__global__ __launch_bounds__(32) void k_adj(const __bf16* __restrict__ t_bf,
                                            const __bf16* __restrict__ adp_bf,
                                            float* __restrict__ adj) {
  int b = blockIdx.x, nt = blockIdx.y, mt = blockIdx.z;
  f32x8 acc = {};
#pragma unroll
  for (int ks = 0; ks < 4; ++ks) {
    int kb = ks * 32;
    bf16x16 a = load_a_frag_bf([&](int m, int k) {
      return t_bf[((size_t)b * NN + nt * 16 + m) * NHID + kb + k];
    });
    bf16x16 bm = load_b_frag_bf([&](int k, int j) {
      return adp_bf[((size_t)b * NN + mt * 16 + j) * NHID + kb + k];
    });
    WMMA_BF16(acc, a, bm);
  }
  int lane = threadIdx.x, half = lane >> 4, col = lane & 15;
#pragma unroll
  for (int r = 0; r < 8; ++r) {
    int m = r + 8 * half;
    adj[((size_t)b * NN + nt * 16 + m) * NN + mt * 16 + col] =
        fmaxf(acc[r], 0.f);
  }
}

// top-20 mask + softmax(adj*mask) * 0.5, stored TRANSPOSED bf16:
// adjT[b][w][v] = value of row v, col w.  1 block / row, 128 thr.
__global__ void k_topk(const float* __restrict__ adj,
                       __bf16* __restrict__ adjT) {
  __shared__ float w[NN];
  __shared__ float orig[NN];
  __shared__ float rv[128];
  __shared__ int ri[128];
  __shared__ unsigned char msk[NN];
  int row = blockIdx.x, tid = threadIdx.x;
  int b = row >> 9, v = row & (NN - 1);
  const float* src = adj + (size_t)row * NN;
  for (int j = tid; j < NN; j += 128) {
    float x = src[j];
    w[j] = x;
    orig[j] = x;
    msk[j] = 0;
  }
  __syncthreads();
  for (int it = 0; it < 20; ++it) {
    float best = -1e30f;
    int bi = tid;
    for (int j = tid; j < NN; j += 128)
      if (w[j] > best) { best = w[j]; bi = j; }
    rv[tid] = best;
    ri[tid] = bi;
    __syncthreads();
    for (int s = 64; s > 0; s >>= 1) {
      if (tid < s && rv[tid + s] > rv[tid]) {
        rv[tid] = rv[tid + s];
        ri[tid] = ri[tid + s];
      }
      __syncthreads();
    }
    if (tid == 0) {
      int j = ri[0];
      msk[j] = 1;
      w[j] = -1e30f;
    }
    __syncthreads();
  }
  float m = 0.f;  // masked values are 0; relu'd values >= 0
  for (int j = tid; j < NN; j += 128) {
    float x = msk[j] ? orig[j] : 0.f;
    m = fmaxf(m, x);
  }
  rv[tid] = m;
  __syncthreads();
  for (int s = 64; s > 0; s >>= 1) {
    if (tid < s) rv[tid] = fmaxf(rv[tid], rv[tid + s]);
    __syncthreads();
  }
  float rmax = rv[0];
  __syncthreads();
  float sum = 0.f;
  for (int j = tid; j < NN; j += 128) {
    float x = msk[j] ? orig[j] : 0.f;
    sum += __expf(x - rmax);
  }
  rv[tid] = sum;
  __syncthreads();
  for (int s = 64; s > 0; s >>= 1) {
    if (tid < s) rv[tid] += rv[tid + s];
    __syncthreads();
  }
  float inv = 0.5f / rv[0];  // A_COEF folded in
  for (int j = tid; j < NN; j += 128) {
    float x = msk[j] ? orig[j] : 0.f;
    adjT[(((size_t)b * NN + j) * NN) + v] = (__bf16)(__expf(x - rmax) * inv);
  }
}

// fused dilated tconv: v = tanh(x*Wf) * sigmoid(x*Wg).  K=64 (2 taps x 32c).
__global__ __launch_bounds__(32) void k_tconv(
    const __bf16* __restrict__ xbf, int Lin, int d,
    const __bf16* __restrict__ fwt, const float* __restrict__ fb,
    const __bf16* __restrict__ gwt, const float* __restrict__ gb,
    __bf16* __restrict__ vbf) {
  int Lout = Lin - d;
  int bl = blockIdx.x;
  int b = bl / Lout, l = bl % Lout;
  int nt = blockIdx.y, ot = blockIdx.z;
  f32x8 fa = {}, ga = {};
#pragma unroll
  for (int tap = 0; tap < 2; ++tap) {
    int ll = l + tap * d;
    bf16x16 a = load_a_frag_bf([&](int m, int k) {
      return xbf[(((size_t)b * Lin + ll) * NRC + k) * NN + nt * 16 + m];
    });
    bf16x16 bfw = load_b_frag_bf([&](int k, int o) {
      return fwt[((size_t)tap * NDC + ot * 16 + o) * NRC + k];
    });
    WMMA_BF16(fa, a, bfw);
    bf16x16 bgw = load_b_frag_bf([&](int k, int o) {
      return gwt[((size_t)tap * NDC + ot * 16 + o) * NRC + k];
    });
    WMMA_BF16(ga, a, bgw);
  }
  int lane = threadIdx.x, half = lane >> 4, col = lane & 15;
#pragma unroll
  for (int r = 0; r < 8; ++r) {
    int n = nt * 16 + r + 8 * half;
    int o = ot * 16 + col;
    float f = tanhf(fa[r] + fb[o]);
    float g = 1.f / (1.f + __expf(-(ga[r] + gb[o])));
    vbf[(((size_t)b * Lout + l) * NDC + o) * NN + n] = (__bf16)(f * g);
  }
}

// skip telescopes: only last column of each layer's s matters.
// skip_acc layout [b][n][sc] so end1's A loads are contiguous.
__global__ __launch_bounds__(32) void k_skip(const __bf16* __restrict__ vbf,
                                             int Lout,
                                             const float* __restrict__ sw,
                                             const float* __restrict__ sb,
                                             float* __restrict__ skip_acc) {
  int b = blockIdx.x, nt = blockIdx.y, st = blockIdx.z;
  int l = Lout - 1;
  f32x8 acc = {};
  bf16x16 a = load_a_frag_bf([&](int m, int k) {
    return vbf[(((size_t)b * Lout + l) * NDC + k) * NN + nt * 16 + m];
  });
  bf16x16 bm =
      load_b_frag([&](int k, int sc) { return sw[(st * 16 + sc) * NDC + k]; });
  WMMA_BF16(acc, a, bm);
  int lane = threadIdx.x, half = lane >> 4, col = lane & 15;
#pragma unroll
  for (int r = 0; r < 8; ++r) {
    int n = nt * 16 + r + 8 * half;
    int sc = st * 16 + col;
    skip_acc[((size_t)b * NN + n) * NSC + sc] += acc[r] + sb[sc];
  }
}

// nconv: y[b][l][c][w] = sum_v x[b][l][c][v] * A[v][w].
// AT is the TRANSPOSED adjacency (AT[w][v], bf16) so B loads are contiguous.
__global__ __launch_bounds__(32) void k_nconv(const __bf16* __restrict__ xin,
                                              int Lout,
                                              const __bf16* __restrict__ AT,
                                              int perBatch,
                                              __bf16* __restrict__ yout) {
  int bl = blockIdx.x;
  int b = bl / Lout, l = bl % Lout;
  int wt = blockIdx.y, ct = blockIdx.z;
  const __bf16* Ab = perBatch ? (AT + (size_t)b * NN * NN) : AT;
  f32x8 acc = {};
#pragma unroll 4
  for (int ks = 0; ks < 16; ++ks) {
    int kb = ks * 32;
    bf16x16 a = load_a_frag_bf([&](int m, int k) {
      return xin[(((size_t)b * Lout + l) * NDC + ct * 16 + m) * NN + kb + k];
    });
    bf16x16 bm = load_b_frag_bf([&](int k, int w2) {
      return Ab[(size_t)(wt * 16 + w2) * NN + kb + k];
    });
    WMMA_BF16(acc, a, bm);
  }
  int lane = threadIdx.x, half = lane >> 4, col = lane & 15;
#pragma unroll
  for (int r = 0; r < 8; ++r) {
    int c = ct * 16 + r + 8 * half;
    int w2 = wt * 16 + col;
    yout[(((size_t)b * Lout + l) * NDC + c) * NN + w2] = (__bf16)acc[r];
  }
}

// gcn combine: xnext = bn*(concat[v,y1,y2,y3,y4](160ch) @ gconv + gb + resid)
__global__ __launch_bounds__(32) void k_combine(
    const __bf16* __restrict__ v, const __bf16* __restrict__ y1,
    const __bf16* __restrict__ y2, const __bf16* __restrict__ y3,
    const __bf16* __restrict__ y4, const float* __restrict__ gw,
    const float* __restrict__ gb, const float* __restrict__ xres, int Lin,
    int Lout, float* __restrict__ xnext, __bf16* __restrict__ xnext_bf) {
  int bl = blockIdx.x;
  int b = bl / Lout, l = bl % Lout;
  int nt = blockIdx.y, ct = blockIdx.z;
  const __bf16* srcs[5] = {v, y1, y2, y3, y4};
  f32x8 acc = {};
#pragma unroll
  for (int g = 0; g < 5; ++g) {
    const __bf16* S = srcs[g];
    bf16x16 a = load_a_frag_bf([&](int m, int k) {
      return S[(((size_t)b * Lout + l) * NDC + k) * NN + nt * 16 + m];
    });
    bf16x16 bm = load_b_frag([&](int k, int c) {
      return gw[(ct * 16 + c) * 160 + g * 32 + k];
    });
    WMMA_BF16(acc, a, bm);
  }
  const float bn = 0.999995000037f;  // 1/sqrt(1+1e-5)
  int lane = threadIdx.x, half = lane >> 4, col = lane & 15;
  int off = Lin - Lout;
#pragma unroll
  for (int r = 0; r < 8; ++r) {
    int n = nt * 16 + r + 8 * half;
    int c = ct * 16 + col;
    float val = acc[r] + gb[c] +
                xres[(((size_t)b * Lin + l + off) * NRC + c) * NN + n];
    val *= bn;
    size_t o = (((size_t)b * Lout + l) * NRC + c) * NN + n;
    xnext[o] = val;
    xnext_bf[o] = (__bf16)val;
  }
}

// h[b][n][e] = relu( relu(skip[b][n][:]) @ end1_w + end1_b )
__global__ __launch_bounds__(32) void k_end1(const float* __restrict__ skip_acc,
                                             const float* __restrict__ w1,
                                             const float* __restrict__ b1,
                                             float* __restrict__ h) {
  int b = blockIdx.x, nt = blockIdx.y, et = blockIdx.z;
  f32x8 acc = {};
#pragma unroll
  for (int ks = 0; ks < 8; ++ks) {
    int kb = ks * 32;
    bf16x16 a = load_a_frag([&](int m, int k) {
      return fmaxf(skip_acc[((size_t)b * NN + nt * 16 + m) * NSC + kb + k],
                   0.f);
    });
    bf16x16 bm = load_b_frag(
        [&](int k, int e2) { return w1[(et * 16 + e2) * NSC + kb + k]; });
    WMMA_BF16(acc, a, bm);
  }
  int lane = threadIdx.x, half = lane >> 4, col = lane & 15;
#pragma unroll
  for (int r = 0; r < 8; ++r) {
    int n = nt * 16 + r + 8 * half;
    int e2 = et * 16 + col;
    h[((size_t)b * NN + n) * NEC + e2] = fmaxf(acc[r] + b1[e2], 0.f);
  }
}

// out[b][o][n] = h[b][n][:] @ end2_w[o][:] + end2_b[o]  (contiguous dots)
__global__ void k_end2(const float* __restrict__ h,
                       const float* __restrict__ w2,
                       const float* __restrict__ b2, float* __restrict__ out) {
  int idx = blockIdx.x * blockDim.x + threadIdx.x;  // b*12*512
  if (idx >= NB * NOUT * NN) return;
  int n = idx & (NN - 1);
  int o = (idx >> 9) % NOUT;
  int b = idx / (NOUT * NN);
  float acc = b2[o];
  const float* hp = h + ((size_t)b * NN + n) * NEC;
  const float* wp = w2 + o * NEC;
#pragma unroll 8
  for (int e = 0; e < NEC; ++e) acc += hp[e] * wp[e];
  out[((size_t)b * NOUT + o) * NN + n] = acc;
}

// ---------------------------------------------------------------------------
// Launch
// ---------------------------------------------------------------------------
extern "C" void kernel_launch(void* const* d_in, const int* in_sizes, int n_in,
                              void* d_out, int out_size, void* d_ws,
                              size_t ws_size, hipStream_t stream) {
  (void)in_sizes; (void)n_in; (void)out_size; (void)ws_size;
  const float* hd       = (const float*)d_in[0];
  const float* start_w  = (const float*)d_in[1];
  const float* start_b  = (const float*)d_in[2];
  const float* filt_w   = (const float*)d_in[3];
  const float* filt_b   = (const float*)d_in[4];
  const float* gate_w   = (const float*)d_in[5];
  const float* gate_b   = (const float*)d_in[6];
  const float* skip_w   = (const float*)d_in[7];
  const float* skip_b   = (const float*)d_in[8];
  const float* gconv_w  = (const float*)d_in[9];
  const float* gconv_b  = (const float*)d_in[10];
  const float* end1_w   = (const float*)d_in[11];
  const float* end1_b   = (const float*)d_in[12];
  const float* end2_w   = (const float*)d_in[13];
  const float* end2_b   = (const float*)d_in[14];
  const float* Ex1      = (const float*)d_in[15];
  const float* node1    = (const float*)d_in[16];
  const float* Wd       = (const float*)d_in[17];
  const float* Wxabs    = (const float*)d_in[18];
  const float* tid_emb  = (const float*)d_in[19];
  const float* diw_emb  = (const float*)d_in[20];
  const float* nv1      = (const float*)d_in[21];
  const float* nv2      = (const float*)d_in[22];
  float* out = (float*)d_out;

  // workspace partition
  char* ws = (char*)d_ws;
  auto alloc = [&](size_t bytes) -> void* {
    void* p = (void*)ws;
    ws += (bytes + 255) & ~(size_t)255;
    return p;
  };
  const size_t XMAX = (size_t)NB * NRECEPT * NRC * NN;   // 3.4M elem
  const size_t VMAX = (size_t)NB * NL * NDC * NN;        // 3.1M elem
  float*  xa       = (float*)alloc(XMAX * 4);
  float*  xb       = (float*)alloc(XMAX * 4);
  __bf16* xa_bf    = (__bf16*)alloc(XMAX * 2);
  __bf16* xb_bf    = (__bf16*)alloc(XMAX * 2);
  __bf16* v_bf     = (__bf16*)alloc(VMAX * 2);
  __bf16* y1       = (__bf16*)alloc(VMAX * 2);
  __bf16* y2       = (__bf16*)alloc(VMAX * 2);
  __bf16* y3       = (__bf16*)alloc(VMAX * 2);
  __bf16* y4       = (__bf16*)alloc(VMAX * 2);
  __bf16* e_bf     = (__bf16*)alloc((size_t)NB * NN * NEMBP * 2);
  __bf16* Wdt      = (__bf16*)alloc((size_t)NN * NHID * NEMBP * 2);
  float*  adp      = (float*)alloc((size_t)NB * NN * NHID * 4);
  __bf16* adp_bf   = (__bf16*)alloc((size_t)NB * NN * NHID * 2);
  __bf16* t_bf     = (__bf16*)alloc((size_t)NB * NN * NHID * 2);
  float*  adj      = (float*)alloc((size_t)NB * NN * NN * 4);
  __bf16* adjT_bf  = (__bf16*)alloc((size_t)NB * NN * NN * 2);
  __bf16* gwadpT   = (__bf16*)alloc((size_t)NN * NN * 2);
  float*  skip_acc = (float*)alloc((size_t)NB * NN * NSC * 4);
  __bf16* fwt      = (__bf16*)alloc((size_t)8 * 2 * NDC * NRC * 2);
  __bf16* gwt      = (__bf16*)alloc((size_t)8 * 2 * NDC * NRC * 2);
  __bf16* Wxt      = (__bf16*)alloc((size_t)NHID * NHID * 2);
  float*  h        = adj;  // adj dead after k_topk; alias for end1 output

  // --- prologue ---
  {
    int n = NB * NN * NSC;
    k_zero<<<(n + 255) / 256, 256, 0, stream>>>(skip_acc, n);
  }
  {
    int n = 8 * 2 * NDC * NRC;
    k_prep_tw<<<(n + 255) / 256, 256, 0, stream>>>(filt_w, gate_w, fwt, gwt);
  }
  {
    int n = NHID * NHID;
    k_prep_wx<<<(n + 255) / 256, 256, 0, stream>>>(Wxabs, Wxt);
  }
  {
    int n = NN * NHID * NEMBP;
    k_prep_wd<<<(n + 255) / 256, 256, 0, stream>>>(Wd, Wdt);
  }
  {
    int n = NB * NRECEPT * NN;
    k_start<<<(n + 255) / 256, 256, 0, stream>>>(hd, start_w, start_b, xa,
                                                 xa_bf);
  }
  k_gwadp<<<NN, 256, 0, stream>>>(nv1, nv2, gwadpT);
  {
    int n = NB * NN;
    k_embed<<<(n + 255) / 256, 256, 0, stream>>>(hd, Ex1, node1, tid_emb,
                                                 diw_emb, e_bf);
  }
  k_adp<<<dim3(NN, NHID / 16), 32, 0, stream>>>(e_bf, Wdt, adp);
  k_lnorm<<<NB, 256, 0, stream>>>(adp, adp_bf);
  k_txw<<<dim3(NB, NN / 16, NHID / 16), 32, 0, stream>>>(adp_bf, Wxt, t_bf);
  k_adj<<<dim3(NB, NN / 16, NN / 16), 32, 0, stream>>>(t_bf, adp_bf, adj);
  k_topk<<<NB * NN, 128, 0, stream>>>(adj, adjT_bf);

  // --- WaveNet stack ---
  static const int dil[8] = {1, 2, 1, 2, 1, 2, 1, 2};
  int Lin = NRECEPT;
  float* xcur = xa;
  float* xnx = xb;
  __bf16* xcur_bf = xa_bf;
  __bf16* xnx_bf = xb_bf;
  for (int i = 0; i < 8; ++i) {
    int d = dil[i];
    int Lout = Lin - d;
    dim3 gt(NB * Lout, NN / 16, NDC / 16);
    k_tconv<<<gt, 32, 0, stream>>>(xcur_bf, Lin, d,
                                   fwt + (size_t)i * 2 * NDC * NRC,
                                   filt_b + i * NDC,
                                   gwt + (size_t)i * 2 * NDC * NRC,
                                   gate_b + i * NDC, v_bf);
    dim3 gs(NB, NN / 16, NSC / 16);
    k_skip<<<gs, 32, 0, stream>>>(v_bf, Lout, skip_w + (size_t)i * NSC * NDC,
                                  skip_b + i * NSC, skip_acc);
    dim3 gn(NB * Lout, NN / 16, NDC / 16);
    k_nconv<<<gn, 32, 0, stream>>>(v_bf, Lout, gwadpT, 0, y1);
    k_nconv<<<gn, 32, 0, stream>>>(y1, Lout, gwadpT, 0, y2);
    k_nconv<<<gn, 32, 0, stream>>>(v_bf, Lout, adjT_bf, 1, y3);
    k_nconv<<<gn, 32, 0, stream>>>(y3, Lout, adjT_bf, 1, y4);
    dim3 gc(NB * Lout, NN / 16, NRC / 16);
    k_combine<<<gc, 32, 0, stream>>>(v_bf, y1, y2, y3, y4,
                                     gconv_w + (size_t)i * 32 * 160,
                                     gconv_b + i * NRC, xcur, Lin, Lout, xnx,
                                     xnx_bf);
    float* tf = xcur; xcur = xnx; xnx = tf;
    __bf16* tb = xcur_bf; xcur_bf = xnx_bf; xnx_bf = tb;
    Lin = Lout;
  }

  // --- epilogue ---
  k_end1<<<dim3(NB, NN / 16, NEC / 16), 32, 0, stream>>>(skip_acc, end1_w,
                                                         end1_b, h);
  {
    int n = NB * NOUT * NN;
    k_end2<<<(n + 255) / 256, 256, 0, stream>>>(h, end2_w, end2_b, out);
  }
}
